// AverageAttn_71184787964284
// MI455X (gfx1250) — matrix operand order repaired and checked by reading, not compiled, not run on hardware
//
#include <hip/hip_runtime.h>
#include <hip/hip_bf16.h>

// ---------------------------------------------------------------------------
// Types
// ---------------------------------------------------------------------------
typedef __bf16 bf16_t;
typedef __attribute__((ext_vector_type(16))) __bf16 v16bf;
typedef __attribute__((ext_vector_type(8)))  float  v8f;
typedef __attribute__((ext_vector_type(4)))  float  f32x4;
typedef __attribute__((ext_vector_type(4)))  int    v4i;

union Frag16 {            // 16 bf16 elements = 8 VGPRs, loaded as two 16B chunks
  v16bf v;
  f32x4 q[2];
};

#define BM   128          // block tile rows
#define BN   128          // block tile cols (G1/G2)
#define GN   64           // block tile cols per gate (G3)
#define BK   32           // K step (matches 16x16x32 bf16 WMMA)
#define LDP  40           // padded LDS row pitch in bf16 (80B -> conflict-free)

// ---------------------------------------------------------------------------
// CDNA5 async global->LDS staging (ASYNCcnt path), guarded with fallback.
// Probe-confirmed signature: (v4i addrspace(1)*, v4i addrspace(3)*, Ii, Ii)
// ---------------------------------------------------------------------------
#if defined(__has_builtin)
#if __has_builtin(__builtin_amdgcn_global_load_async_to_lds_b128)
#define ASYNC_LDS 1
#endif
#endif

__device__ __forceinline__ void stage16B(const bf16_t* g, bf16_t* l) {
#if defined(ASYNC_LDS)
  __builtin_amdgcn_global_load_async_to_lds_b128(
      (__attribute__((address_space(1))) v4i*)g,
      (__attribute__((address_space(3))) v4i*)l,
      /*offset=*/0, /*cpol=*/0);
#else
  *(f32x4*)l = *(const f32x4*)g;
#endif
}

template <int N>
__device__ __forceinline__ void wait_async() {
#if defined(ASYNC_LDS)
#if __has_builtin(__builtin_amdgcn_s_wait_asynccnt)
  __builtin_amdgcn_s_wait_asynccnt(N);
#else
  asm volatile("s_wait_asynccnt %0" ::"i"(N) : "memory");
#endif
#endif
}

__device__ __forceinline__ v8f wmma_bf16(v16bf a, v16bf b, v8f c) {
  // D = A(16x32) * B(32x16) + C, f32 accumulate
  return __builtin_amdgcn_wmma_f32_16x16x32_bf16(
      /*neg_a=*/false, a, /*neg_b=*/false, b,
      /*c_mod=*/(short)0, c, /*reuse_a=*/false, /*reuse_b=*/false);
}

__device__ __forceinline__ float sigmoidf(float x) {
  return 1.0f / (1.0f + __expf(-x));
}

// ---------------------------------------------------------------------------
// f32 -> bf16 convert (elementwise)
// ---------------------------------------------------------------------------
__global__ void cvt_bf16_kernel(const float* __restrict__ src,
                                bf16_t* __restrict__ dst, int n) {
  int i = blockIdx.x * blockDim.x + threadIdx.x;
  if (i < n) dst[i] = (bf16_t)src[i];
}

// ---------------------------------------------------------------------------
// Tiled transpose + convert: src R x C (f32, row-major) -> dst C x R (bf16).
// dst[c*R + r] = src[r*C + c]. Coalesced reads and writes via 32x32 LDS tile.
// ---------------------------------------------------------------------------
__global__ __launch_bounds__(256)
void transpose_cvt_kernel(const float* __restrict__ src,
                          bf16_t* __restrict__ dst, int R, int C) {
  __shared__ float tile[32][33];
  const int c0 = blockIdx.x * 32;
  const int r0 = blockIdx.y * 32;
  const int tx = threadIdx.x & 31;
  const int ty = threadIdx.x >> 5;            // 8 rows per pass
#pragma unroll
  for (int p = 0; p < 4; ++p) {
    int r = r0 + ty + p * 8;
    tile[ty + p * 8][tx] = src[(size_t)r * C + c0 + tx];
  }
  __syncthreads();
#pragma unroll
  for (int p = 0; p < 4; ++p) {
    int c = c0 + ty + p * 8;
    dst[(size_t)c * R + r0 + tx] = (bf16_t)tile[tx][ty + p * 8];
  }
}

// ---------------------------------------------------------------------------
// Causal running mean along S. One thread per (b, d) column; loads are
// coalesced across d within a wave. Writes avg as bf16 (feeds GEMM1 only).
// ---------------------------------------------------------------------------
__global__ void cumavg_kernel(const float* __restrict__ iV,
                              bf16_t* __restrict__ avg_bf, int S, int D) {
  int t = blockIdx.x * blockDim.x + threadIdx.x;   // t in [0, B*D)
  int b = t / D;
  int d = t - b * D;
  const float* src = iV + (size_t)b * S * D + d;
  bf16_t* dst = avg_bf + (size_t)b * S * D + d;
  float acc = 0.0f;
  for (int s = 0; s < S; ++s) {
    acc += src[(size_t)s * D];
    float avg = acc / (float)(s + 1);
    dst[(size_t)s * D] = (bf16_t)avg;
  }
}

// ---------------------------------------------------------------------------
// C[M,N] = A[M,K] @ Bt[N,K]^T + bias, bf16 inputs / f32 accumulate via WMMA.
// Bt is the PRE-TRANSPOSED weight (N x K row-major) so both A and B tiles are
// contiguous row chunks -> staged with async global->LDS b128, double-buffered.
// MODE 0: outBf = bf16(relu(acc + bias))
// MODE 1: outF  = acc + bias (f32),  outBf = bf16(same)
// Block: 256 threads (8 waves), tile 128x128, waves 4(M) x 2(N),
// each wave: 2x4 tiles of 16x16 -> 8 accumulators.
// Async instrs per wave per tile: A=2, B=2 -> threshold 4.
// ---------------------------------------------------------------------------
template <int MODE>
__global__ __launch_bounds__(256)
void gemm_bias_kernel(const bf16_t* __restrict__ A,
                      const bf16_t* __restrict__ Bt,
                      const float* __restrict__ bias,
                      bf16_t* __restrict__ outBf,
                      float* __restrict__ outF,
                      int M, int N, int K) {
  __shared__ __attribute__((aligned(16))) bf16_t As[2][BM * LDP];  // [buf][row][k]
  __shared__ __attribute__((aligned(16))) bf16_t Bs[2][BN * LDP];  // [buf][col][k]

  const int tid   = threadIdx.x;
  const int lane  = tid & 31;
  const int wid   = tid >> 5;
  const int waveM = wid >> 1;       // 0..3
  const int waveN = wid & 1;        // 0..1
  const int mlan  = lane & 15;
  const int half  = lane >> 4;

  const int bm0 = blockIdx.y * BM;
  const int bn0 = blockIdx.x * BN;

  v8f acc[2][4];
#pragma unroll
  for (int i = 0; i < 2; ++i)
#pragma unroll
    for (int j = 0; j < 4; ++j) acc[i][j] = (v8f)0.0f;

  const int rowA = tid >> 2;          // 0..63 (+64 on pass 1)
  const int chA  = tid & 3;           // 16B chunk within 64B row

  // preload tile 0 into buffer 0
#pragma unroll
  for (int p = 0; p < 2; ++p) {
    int row = rowA + p * 64;
    stage16B(&A[(size_t)(bm0 + row) * K + chA * 8], &As[0][row * LDP + chA * 8]);
  }
#pragma unroll
  for (int p = 0; p < 2; ++p) {
    int n = rowA + p * 64;
    stage16B(&Bt[(size_t)(bn0 + n) * K + chA * 8], &Bs[0][n * LDP + chA * 8]);
  }

  for (int k0 = 0; k0 < K; k0 += BK) {
    const int cur = (k0 / BK) & 1;
    const bool has_next = (k0 + BK) < K;
    if (has_next) {
      const int nxt = cur ^ 1;
      const int kn = k0 + BK;
#pragma unroll
      for (int p = 0; p < 2; ++p) {
        int row = rowA + p * 64;
        stage16B(&A[(size_t)(bm0 + row) * K + kn + chA * 8],
                 &As[nxt][row * LDP + chA * 8]);
      }
#pragma unroll
      for (int p = 0; p < 2; ++p) {
        int n = rowA + p * 64;
        stage16B(&Bt[(size_t)(bn0 + n) * K + kn + chA * 8],
                 &Bs[nxt][n * LDP + chA * 8]);
      }
      wait_async<4>();     // current tile's 4 async instrs complete
    } else {
      wait_async<0>();
    }
    __syncthreads();       // all waves' current tile resident

    // ---- fragments ----
    Frag16 af[2];
#pragma unroll
    for (int i = 0; i < 2; ++i) {
      int r = waveM * 32 + i * 16 + mlan;
      af[i].q[0] = *(const f32x4*)&As[cur][r * LDP + half * 16];      // K 16h..
      af[i].q[1] = *(const f32x4*)&As[cur][r * LDP + half * 16 + 8];  // K 16h+8..
    }
    Frag16 bfr[4];
#pragma unroll
    for (int j = 0; j < 4; ++j) {
      int c = waveN * 64 + j * 16 + mlan;
      bfr[j].q[0] = *(const f32x4*)&Bs[cur][c * LDP + half * 16];
      bfr[j].q[1] = *(const f32x4*)&Bs[cur][c * LDP + half * 16 + 8];
    }
#pragma unroll
    for (int i = 0; i < 2; ++i)
#pragma unroll
      for (int j = 0; j < 4; ++j)
        acc[i][j] = wmma_bf16(af[i].v, bfr[j].v, acc[i][j]);
    __syncthreads();       // done reading cur -> safe to refill next iter
  }

  // ---- epilogue ----
#pragma unroll
  for (int i = 0; i < 2; ++i)
#pragma unroll
    for (int j = 0; j < 4; ++j) {
      int col = bn0 + waveN * 64 + j * 16 + mlan;
      float bv = bias[col];
#pragma unroll
      for (int r = 0; r < 8; ++r) {
        int row = bm0 + waveM * 32 + i * 16 + r + 8 * half;
        float x = acc[i][j][r] + bv;
        if (MODE == 0) {
          x = fmaxf(x, 0.0f);
          outBf[(size_t)row * N + col] = (bf16_t)x;
        } else {
          outF[(size_t)row * N + col] = x;
          outBf[(size_t)row * N + col] = (bf16_t)x;
        }
      }
    }
}

// ---------------------------------------------------------------------------
// Gate GEMM + fused output.
// o[:, 0:2048] = [iQ | h] @ Wg + bg ; out = sig(o_i)*iQ + sig(o_f)*h
// Wgt is Wg pre-transposed: [2048 n][2048 k]. Block: rows [bm0,+128), gate
// cols [n0,+64) for BOTH gates (i: n0, f: n0+1024), K = 2048.
// Async instrs per wave per tile: A=2, Bi=1, Bf=1 -> threshold 4.
// ---------------------------------------------------------------------------
__global__ __launch_bounds__(256)
void gate_gemm_kernel(const bf16_t* __restrict__ iQb,
                      const bf16_t* __restrict__ hb,
                      const bf16_t* __restrict__ Wgt,  // [2048 n][2048 k]
                      const float* __restrict__ bg,    // [2048]
                      const float* __restrict__ iQf,   // original f32 iQ
                      const float* __restrict__ hf,    // f32 h
                      float* __restrict__ out, int M) {
  const int D = 1024, K = 2048;
  __shared__ __attribute__((aligned(16))) bf16_t As[2][BM * LDP];
  __shared__ __attribute__((aligned(16))) bf16_t Bis[2][GN * LDP];
  __shared__ __attribute__((aligned(16))) bf16_t Bfs[2][GN * LDP];

  const int tid   = threadIdx.x;
  const int lane  = tid & 31;
  const int wid   = tid >> 5;
  const int waveM = wid >> 1;       // 0..3 -> 32 rows each
  const int waveN = wid & 1;        // 0..1 -> 32 cols each
  const int mlan  = lane & 15;
  const int half  = lane >> 4;

  const int bm0 = blockIdx.y * BM;
  const int n0  = blockIdx.x * GN;

  v8f acci[2][2], accf[2][2];
#pragma unroll
  for (int i = 0; i < 2; ++i)
#pragma unroll
    for (int j = 0; j < 2; ++j) { acci[i][j] = (v8f)0.0f; accf[i][j] = (v8f)0.0f; }

  const int rowA = tid >> 2;
  const int chA  = tid & 3;
  const int nB   = tid >> 2;        // 0..63 (one pass for 64-row B tiles)

  // stage tile at k-offset k0n into buffer buf
  auto stage_tile = [&](int k0n, int buf) {
    const bf16_t* Asrc = (k0n < D) ? iQb : hb;
    const int kk = k0n & (D - 1);
#pragma unroll
    for (int p = 0; p < 2; ++p) {
      int row = rowA + p * 64;
      stage16B(&Asrc[(size_t)(bm0 + row) * D + kk + chA * 8],
               &As[buf][row * LDP + chA * 8]);
    }
    stage16B(&Wgt[(size_t)(n0 + nB) * K + k0n + chA * 8],
             &Bis[buf][nB * LDP + chA * 8]);
    stage16B(&Wgt[(size_t)(D + n0 + nB) * K + k0n + chA * 8],
             &Bfs[buf][nB * LDP + chA * 8]);
  };

  stage_tile(0, 0);

  for (int k0 = 0; k0 < K; k0 += BK) {
    const int cur = (k0 / BK) & 1;
    const bool has_next = (k0 + BK) < K;
    if (has_next) {
      stage_tile(k0 + BK, cur ^ 1);
      wait_async<4>();
    } else {
      wait_async<0>();
    }
    __syncthreads();

    Frag16 af[2];
#pragma unroll
    for (int i = 0; i < 2; ++i) {
      int r = waveM * 32 + i * 16 + mlan;
      af[i].q[0] = *(const f32x4*)&As[cur][r * LDP + half * 16];
      af[i].q[1] = *(const f32x4*)&As[cur][r * LDP + half * 16 + 8];
    }
    Frag16 bi[2], bff[2];
#pragma unroll
    for (int j = 0; j < 2; ++j) {
      int c = waveN * 32 + j * 16 + mlan;
      bi[j].q[0]  = *(const f32x4*)&Bis[cur][c * LDP + half * 16];
      bi[j].q[1]  = *(const f32x4*)&Bis[cur][c * LDP + half * 16 + 8];
      bff[j].q[0] = *(const f32x4*)&Bfs[cur][c * LDP + half * 16];
      bff[j].q[1] = *(const f32x4*)&Bfs[cur][c * LDP + half * 16 + 8];
    }
#pragma unroll
    for (int i = 0; i < 2; ++i)
#pragma unroll
      for (int j = 0; j < 2; ++j) {
        acci[i][j] = wmma_bf16(af[i].v, bi[j].v, acci[i][j]);
        accf[i][j] = wmma_bf16(af[i].v, bff[j].v, accf[i][j]);
      }
    __syncthreads();
  }

  // ---- fused sigmoid gating epilogue ----
#pragma unroll
  for (int i = 0; i < 2; ++i)
#pragma unroll
    for (int j = 0; j < 2; ++j) {
      int col = n0 + waveN * 32 + j * 16 + mlan;
      float bgi = bg[col];
      float bgf = bg[D + col];
#pragma unroll
      for (int r = 0; r < 8; ++r) {
        int row = bm0 + waveM * 32 + i * 16 + r + 8 * half;
        size_t idx = (size_t)row * D + col;
        float gi = sigmoidf(acci[i][j][r] + bgi);
        float gf = sigmoidf(accf[i][j][r] + bgf);
        out[idx] = gi * iQf[idx] + gf * hf[idx];
      }
    }
}

// ---------------------------------------------------------------------------
// Launch
// ---------------------------------------------------------------------------
extern "C" void kernel_launch(void* const* d_in, const int* in_sizes, int n_in,
                              void* d_out, int out_size, void* d_ws, size_t ws_size,
                              hipStream_t stream) {
  const float* iQ = (const float*)d_in[0];
  const float* iV = (const float*)d_in[1];
  const float* W1 = (const float*)d_in[2];
  const float* b1 = (const float*)d_in[3];
  const float* W2 = (const float*)d_in[4];
  const float* b2 = (const float*)d_in[5];
  const float* Wg = (const float*)d_in[6];
  const float* bg = (const float*)d_in[7];
  float* out = (float*)d_out;

  const int B = 4, S = 4096, D = 1024, H = 1024;
  const int MR = B * S;                         // 16384 rows

  char* ws = (char*)d_ws;
  const size_t MB = 1024ull * 1024ull;
  bf16_t* iQb  = (bf16_t*)(ws + 0 * MB);        // 32 MB
  bf16_t* avgb = (bf16_t*)(ws + 32 * MB);       // 32 MB (reused as h_bf)
  bf16_t* x1b  = (bf16_t*)(ws + 64 * MB);       // 32 MB
  float*  hf   = (float*)(ws + 96 * MB);        // 64 MB
  bf16_t* w1t  = (bf16_t*)(ws + 160 * MB);      // 2 MB  (H x D, transposed)
  bf16_t* w2t  = (bf16_t*)(ws + 162 * MB);      // 2 MB  (D x H, transposed)
  bf16_t* wgt  = (bf16_t*)(ws + 164 * MB);      // 8 MB  (2D x 2D, transposed)
  bf16_t* hb   = avgb;                          // avg dead after GEMM1

  // converts (weights transposed so GEMM B tiles are contiguous rows)
  cvt_bf16_kernel<<<(MR * D) / 256, 256, 0, stream>>>(iQ, iQb, MR * D);
  transpose_cvt_kernel<<<dim3(H / 32, D / 32), 256, 0, stream>>>(W1, w1t, D, H);
  transpose_cvt_kernel<<<dim3(D / 32, H / 32), 256, 0, stream>>>(W2, w2t, H, D);
  transpose_cvt_kernel<<<dim3((2 * D) / 32, (2 * D) / 32), 256, 0, stream>>>(
      Wg, wgt, 2 * D, 2 * D);

  // causal running mean -> avg (bf16)
  cumavg_kernel<<<(B * D) / 256, 256, 0, stream>>>(iV, avgb, S, D);

  // h = relu(avg@W1+b1)@W2+b2
  gemm_bias_kernel<0><<<dim3(H / BN, MR / BM), 256, 0, stream>>>(
      avgb, w1t, b1, x1b, nullptr, MR, H, D);
  gemm_bias_kernel<1><<<dim3(D / BN, MR / BM), 256, 0, stream>>>(
      x1b, w2t, b2, hb, hf, MR, D, H);

  // gates + output
  gate_gemm_kernel<<<dim3(D / GN, MR / BM), 256, 0, stream>>>(
      iQb, hb, wgt, bg, iQ, hf, out, MR);
}
